// DinoVisionTransformer_51419348468125
// MI455X (gfx1250) — matrix operand order repaired and checked by reading, not compiled, not run on hardware
//
#include <hip/hip_runtime.h>

// ---------------------------------------------------------------------------
// Types for CDNA5 WMMA (wave32, 16x16x32 bf16 -> f32)
// ---------------------------------------------------------------------------
typedef __attribute__((ext_vector_type(16))) __bf16 v16bf;
typedef __attribute__((ext_vector_type(8)))  __bf16 v8bf;
typedef __attribute__((ext_vector_type(8)))  float  v8f;

union AFrag { v16bf v; v8bf h[2]; };

// Model constants
#define NBATCH 32
#define NLAYER 12
#define NHEAD  12
#define HDIM   64
#define DMODEL 768      // D (Lorentz dim), spatial = 767
#define DFFN   2048
#define NTOK   197
#define TTOK   (NBATCH*NTOK)       // 6304
#define NPADK  224                 // 197 padded to mult of 32 (attn K dim)
#define EDIM   96                  // 64+1 head dim padded to mult of 32
#define BHEADS (NBATCH*NHEAD)      // 384

// GEMM tiling: 256 threads = 8 waves (2 x 4), each wave computes 64x64.
#define BM 128
#define BN 256
#define BKK 32
#define LSTR 40                    // LDS row stride (bf16): 80B, 16B aligned
#define ASZ (BM*LSTR)              // one A buffer (elems)
#define BSZ (BN*LSTR)              // one B buffer (elems)

// ---------------------------------------------------------------------------
// Batched GEMM: C(f32)[M][N] = A(f32)[M][K] x B^T, where Bt is bf16 [N][K]
// row-major (all weights are packed transposed; attention operands are built
// in this layout directly). K must be a multiple of 32 and zero-padded in
// both operands. Double-buffered LDS, one barrier per K-step, global loads
// for tile k+1 overlap the 16 WMMAs of tile k.
// ---------------------------------------------------------------------------
__global__ __launch_bounds__(256)
void gemm_wmma_kernel(const float* __restrict__ A, const __bf16* __restrict__ Bt,
                      float* __restrict__ C,
                      int Mact, int Nact, int K,
                      int lda, int ldb, int ldc,
                      long sA, long sB, long sC)
{
    __shared__ __bf16 As[2 * ASZ];
    __shared__ __bf16 Bs[2 * BSZ];

    A  += (long)blockIdx.z * sA;
    Bt += (long)blockIdx.z * sB;
    C  += (long)blockIdx.z * sC;

    const int m0   = blockIdx.y * BM;
    const int n0   = blockIdx.x * BN;
    const int tid  = threadIdx.x;
    const int lane = tid & 31;
    const int l16  = lane & 15;
    const int kh   = lane >> 4;      // half-wave selector (ISA 16-bit layouts)
    const int wid  = tid >> 5;       // 8 waves
    const int wm   = wid & 1;        // wave row (2 x 64)
    const int wn   = wid >> 1;       // wave col (4 x 64)

    v8f acc[4][4];
    #pragma unroll
    for (int i = 0; i < 4; ++i)
        #pragma unroll
        for (int j = 0; j < 4; ++j)
            #pragma unroll
            for (int e = 0; e < 8; ++e) acc[i][j][e] = 0.f;

    // staging assignments
    const int ar = tid >> 1;             // A: 2 threads/row, 16 f32 each
    const int ah = (tid & 1) * 16;
    const bool aok = (m0 + ar) < Mact;
    const bool bok = (n0 + tid) < Nact;  // B: 1 thread/row, 32 bf16 each
    const float*  Aptr = A  + (long)(m0 + ar) * lda + ah;
    const __bf16* Bptr = Bt + (long)(n0 + tid) * ldb;

    // zero OOB rows of both LDS buffers once (never rewritten below)
    {
        v8bf z;
        #pragma unroll
        for (int i = 0; i < 8; ++i) z[i] = (__bf16)0.f;
        if (!aok) {
            #pragma unroll
            for (int s = 0; s < 2; ++s) {
                __bf16* d = &As[s * ASZ + ar * LSTR + ah];
                *(v8bf*)d = z; *(v8bf*)(d + 8) = z;
            }
        }
        if (!bok) {
            #pragma unroll
            for (int s = 0; s < 2; ++s) {
                __bf16* d = &Bs[s * BSZ + tid * LSTR];
                *(v8bf*)d = z; *(v8bf*)(d + 8) = z;
                *(v8bf*)(d + 16) = z; *(v8bf*)(d + 24) = z;
            }
        }
    }

    float4 fa0, fa1, fa2, fa3;
    v8bf   rb0, rb1, rb2, rb3;

    // preload tile 0
    if (aok) {
        const float4* s = (const float4*)Aptr;
        fa0 = s[0]; fa1 = s[1]; fa2 = s[2]; fa3 = s[3];
    }
    if (bok) {
        const v8bf* s = (const v8bf*)Bptr;
        rb0 = s[0]; rb1 = s[1]; rb2 = s[2]; rb3 = s[3];
    }
    // store tile 0 into buffer 0
    if (aok) {
        v8bf o0, o1;
        o0[0]=(__bf16)fa0.x; o0[1]=(__bf16)fa0.y; o0[2]=(__bf16)fa0.z; o0[3]=(__bf16)fa0.w;
        o0[4]=(__bf16)fa1.x; o0[5]=(__bf16)fa1.y; o0[6]=(__bf16)fa1.z; o0[7]=(__bf16)fa1.w;
        o1[0]=(__bf16)fa2.x; o1[1]=(__bf16)fa2.y; o1[2]=(__bf16)fa2.z; o1[3]=(__bf16)fa2.w;
        o1[4]=(__bf16)fa3.x; o1[5]=(__bf16)fa3.y; o1[6]=(__bf16)fa3.z; o1[7]=(__bf16)fa3.w;
        __bf16* d = &As[ar * LSTR + ah];
        *(v8bf*)d = o0; *(v8bf*)(d + 8) = o1;
    }
    if (bok) {
        __bf16* d = &Bs[tid * LSTR];
        *(v8bf*)d = rb0; *(v8bf*)(d + 8) = rb1;
        *(v8bf*)(d + 16) = rb2; *(v8bf*)(d + 24) = rb3;
    }
    __syncthreads();

    const int KT = K / BKK;
    for (int kt = 0; kt < KT; ++kt) {
        const int cur = kt & 1;
        const bool more = (kt + 1) < KT;

        // issue global loads for tile kt+1 (overlaps WMMAs below)
        if (more) {
            if (aok) {
                const float4* s = (const float4*)(Aptr + (long)(kt + 1) * BKK);
                if (kt + 2 < KT)
                    __builtin_prefetch(Aptr + (long)(kt + 2) * BKK, 0, 1);
                fa0 = s[0]; fa1 = s[1]; fa2 = s[2]; fa3 = s[3];
            }
            if (bok) {
                const v8bf* s = (const v8bf*)(Bptr + (long)(kt + 1) * BKK);
                rb0 = s[0]; rb1 = s[1]; rb2 = s[2]; rb3 = s[3];
            }
        }

        // fragments per ISA 16-bit layouts + 16 WMMAs
        const __bf16* Ab = &As[cur * ASZ];
        const __bf16* Bb = &Bs[cur * BSZ];
        AFrag afr[4], bfr[4];
        #pragma unroll
        for (int mt = 0; mt < 4; ++mt) {
            int r = wm * 64 + mt * 16 + l16;
            afr[mt].h[0] = *(const v8bf*)&Ab[r * LSTR + kh * 8];
            afr[mt].h[1] = *(const v8bf*)&Ab[r * LSTR + 16 + kh * 8];
        }
        #pragma unroll
        for (int nt = 0; nt < 4; ++nt) {
            int c = wn * 64 + nt * 16 + l16;
            bfr[nt].h[0] = *(const v8bf*)&Bb[c * LSTR + kh * 8];
            bfr[nt].h[1] = *(const v8bf*)&Bb[c * LSTR + 16 + kh * 8];
        }
        #pragma unroll
        for (int mt = 0; mt < 4; ++mt)
            #pragma unroll
            for (int nt = 0; nt < 4; ++nt)
                acc[mt][nt] = __builtin_amdgcn_wmma_f32_16x16x32_bf16(
                    false, afr[mt].v, false, bfr[nt].v,
                    (short)0, acc[mt][nt], false, false);

        // stage tile kt+1 into the other buffer
        if (more) {
            if (aok) {
                v8bf o0, o1;
                o0[0]=(__bf16)fa0.x; o0[1]=(__bf16)fa0.y; o0[2]=(__bf16)fa0.z; o0[3]=(__bf16)fa0.w;
                o0[4]=(__bf16)fa1.x; o0[5]=(__bf16)fa1.y; o0[6]=(__bf16)fa1.z; o0[7]=(__bf16)fa1.w;
                o1[0]=(__bf16)fa2.x; o1[1]=(__bf16)fa2.y; o1[2]=(__bf16)fa2.z; o1[3]=(__bf16)fa2.w;
                o1[4]=(__bf16)fa3.x; o1[5]=(__bf16)fa3.y; o1[6]=(__bf16)fa3.z; o1[7]=(__bf16)fa3.w;
                __bf16* d = &As[(cur ^ 1) * ASZ + ar * LSTR + ah];
                *(v8bf*)d = o0; *(v8bf*)(d + 8) = o1;
            }
            if (bok) {
                __bf16* d = &Bs[(cur ^ 1) * BSZ + tid * LSTR];
                *(v8bf*)d = rb0; *(v8bf*)(d + 8) = rb1;
                *(v8bf*)(d + 16) = rb2; *(v8bf*)(d + 24) = rb3;
            }
        }
        __syncthreads();
    }

    // ---- store C (16x16 f32 C/D layout: lane half selects M base) ----
    #pragma unroll
    for (int mt = 0; mt < 4; ++mt)
        #pragma unroll
        for (int nt = 0; nt < 4; ++nt) {
            int n = n0 + wn * 64 + nt * 16 + l16;
            int mbase = m0 + wm * 64 + mt * 16 + kh * 8;
            if (n < Nact) {
                #pragma unroll
                for (int v = 0; v < 8; ++v) {
                    int m = mbase + v;
                    if (m < Mact) C[(long)m * ldc + n] = acc[mt][nt][v];
                }
            }
        }
}

// ---------------------------------------------------------------------------
// f32 [K][N] -> bf16 transposed [Np][Kp] with zero padding
// ---------------------------------------------------------------------------
__global__ void pack_wT_kernel(const float* __restrict__ src, __bf16* __restrict__ dst,
                               int K, int N, int Kp, int Np, int dstStride, long total)
{
    long g = (long)blockIdx.x * blockDim.x + threadIdx.x;
    if (g >= total) return;
    int k = (int)(g % Kp);
    int n = (int)(g / Kp);
    float v = (k < K && n < N) ? src[(long)k * N + n] : 0.f;
    dst[(long)n * dstStride + k] = (__bf16)v;
}

// ---------------------------------------------------------------------------
// Block reductions (256 threads)
// ---------------------------------------------------------------------------
__device__ __forceinline__ float block_sum(float v, float* sbuf)
{
    #pragma unroll
    for (int off = 16; off > 0; off >>= 1) v += __shfl_down(v, off, 32);
    int lane = threadIdx.x & 31, wid = threadIdx.x >> 5;
    if (lane == 0) sbuf[wid] = v;
    __syncthreads();
    float r = sbuf[0] + sbuf[1] + sbuf[2] + sbuf[3] + sbuf[4] + sbuf[5] + sbuf[6] + sbuf[7];
    __syncthreads();
    return r;
}

__device__ __forceinline__ float block_max(float v, float* sbuf)
{
    #pragma unroll
    for (int off = 16; off > 0; off >>= 1) v = fmaxf(v, __shfl_down(v, off, 32));
    int lane = threadIdx.x & 31, wid = threadIdx.x >> 5;
    if (lane == 0) sbuf[wid] = v;
    __syncthreads();
    float r = fmaxf(fmaxf(fmaxf(sbuf[0], sbuf[1]), fmaxf(sbuf[2], sbuf[3])),
                    fmaxf(fmaxf(sbuf[4], sbuf[5]), fmaxf(sbuf[6], sbuf[7])));
    __syncthreads();
    return r;
}

// ---------------------------------------------------------------------------
// im2col for the lorentz patch conv: row=(b,py,px), col=(iy,ix,c), c0 = time
// ---------------------------------------------------------------------------
__global__ __launch_bounds__(256)
void build_patches_kernel(const float* __restrict__ x, float* __restrict__ Ap)
{
    int row = blockIdx.x;                 // 0..6271
    int b = row / 196, p = row % 196;
    int py = p / 14, px = p % 14;
    int pix = threadIdx.x;                // 0..255
    int iy = pix >> 4, ix = pix & 15;
    int Y = py * 16 + iy, X = px * 16 + ix;
    const float* xb = x + (long)b * 3 * 224 * 224;
    float x0 = xb[(long)(0 * 224 + Y) * 224 + X];
    float x1 = xb[(long)(1 * 224 + Y) * 224 + X];
    float x2 = xb[(long)(2 * 224 + Y) * 224 + X];
    float tt = sqrtf(fmaxf(x0 * x0 + x1 * x1 + x2 * x2 + 1.f, 1e-6f));
    float* dst = Ap + (long)row * 1024 + pix * 4;
    dst[0] = tt; dst[1] = x0; dst[2] = x1; dst[3] = x2;
}

// to_lorentz over feat rows (and cls) -> tok[T][768], tok[0]=time
__global__ __launch_bounds__(256)
void build_tok_kernel(const float* __restrict__ feat, const float* __restrict__ cls_s,
                      float* __restrict__ tok)
{
    __shared__ float sbuf[8];
    int t = blockIdx.x;                   // 0..6303
    int b = t / NTOK, n = t % NTOK;
    const float* src = (n == 0) ? cls_s : (feat + (long)(b * 196 + n - 1) * 768);
    float local[3]; float ss = 0.f;
    #pragma unroll
    for (int j = 0; j < 3; ++j) {
        int i = threadIdx.x + j * 256;
        float v = (i < 767) ? src[i] : 0.f;
        local[j] = v; ss += v * v;
    }
    float tm = sqrtf(fmaxf(block_sum(ss, sbuf) + 1.f, 1e-6f));
    float* dst = tok + (long)t * 768;
    if (threadIdx.x == 0) dst[0] = tm;
    #pragma unroll
    for (int j = 0; j < 3; ++j) {
        int i = threadIdx.x + j * 256;
        if (i < 767) dst[1 + i] = local[j];
    }
}

// Lorentz layer norm on spatial part.
// spatial_only=1: write normalized xs into out[T][768] (col 767 = 0) for GEMM A.
// spatial_only=0: write [time, xs] (final LN -> d_out).
__global__ __launch_bounds__(256)
void lorentz_ln_kernel(const float* __restrict__ tok, float* __restrict__ out,
                       const float* __restrict__ g, const float* __restrict__ b,
                       int spatial_only)
{
    __shared__ float sbuf[8];
    int t = blockIdx.x;
    const float* x = tok + (long)t * 768 + 1;
    float local[3]; float s = 0.f;
    #pragma unroll
    for (int j = 0; j < 3; ++j) {
        int i = threadIdx.x + j * 256;
        float v = (i < 767) ? x[i] : 0.f;
        local[j] = v; s += v;
    }
    float mu = block_sum(s, sbuf) * (1.f / 767.f);
    float s2 = 0.f;
    #pragma unroll
    for (int j = 0; j < 3; ++j) {
        int i = threadIdx.x + j * 256;
        if (i < 767) { float d = local[j] - mu; s2 += d * d; }
    }
    float var = block_sum(s2, sbuf) * (1.f / 767.f);
    float inv = rsqrtf(var + 1e-6f);
    float y[3]; float sy2 = 0.f;
    #pragma unroll
    for (int j = 0; j < 3; ++j) {
        int i = threadIdx.x + j * 256;
        if (i < 767) {
            y[j] = (local[j] - mu) * inv * g[i] + b[i];
            sy2 += y[j] * y[j];
        } else y[j] = 0.f;
    }
    if (spatial_only) {
        #pragma unroll
        for (int j = 0; j < 3; ++j) {
            int i = threadIdx.x + j * 256;
            if (i < 767) out[(long)t * 768 + i] = y[j];
            else if (i == 767) out[(long)t * 768 + 767] = 0.f;
        }
    } else {
        float tt = sqrtf(fmaxf(block_sum(sy2, sbuf) + 1.f, 1e-6f));
        if (threadIdx.x == 0) out[(long)t * 768] = tt;
        #pragma unroll
        for (int j = 0; j < 3; ++j) {
            int i = threadIdx.x + j * 256;
            if (i < 767) out[(long)t * 768 + 1 + i] = y[j];
        }
    }
}

// tok = project(tok + w * to_lorentz(s_in))
__global__ __launch_bounds__(256)
void resadd_project_kernel(float* __restrict__ tok, const float* __restrict__ s_in,
                           const float* __restrict__ wptr, int widx)
{
    __shared__ float sbuf[8];
    int t = blockIdx.x;
    float w = wptr[widx];
    const float* s = s_in + (long)t * 768;
    float* z = tok + (long)t * 768;
    float local[3]; float ss = 0.f;
    #pragma unroll
    for (int j = 0; j < 3; ++j) {
        int i = threadIdx.x + j * 256;
        float v = (i < 767) ? s[i] : 0.f;
        local[j] = v; ss += v * v;
    }
    float ta = sqrtf(fmaxf(block_sum(ss, sbuf) + 1.f, 1e-6f));
    float z0 = z[0] + w * ta;
    float zl[3]; float zs2 = 0.f;
    #pragma unroll
    for (int j = 0; j < 3; ++j) {
        int i = threadIdx.x + j * 256;
        if (i < 767) { zl[j] = z[1 + i] + w * local[j]; zs2 += zl[j] * zl[j]; }
        else zl[j] = 0.f;
    }
    float sum2 = block_sum(zs2, sbuf);
    float f = rsqrtf(fmaxf(z0 * z0 - sum2, 1e-6f));   // sqrt(C)=1
    if (threadIdx.x == 0) z[0] = z0 * f;
    #pragma unroll
    for (int j = 0; j < 3; ++j) {
        int i = threadIdx.x + j * 256;
        if (i < 767) z[1 + i] = zl[j] * f;
    }
}

// Split qkv, apply RoPE, emit:
//   qe  f32  [bh][197][96]  = [q_rot, qt, 0...]          (GEMM A operand)
//   ke  bf16 [bh][197][96]  = [k_rot, -kt, 0...]         (B^T for scores)
//   vfT bf16 [bh][96][224]  : vfT[e][m] = vf[m][e], rows m>=197 zero (B^T for mid)
__global__ void build_qkevf_kernel(const float* __restrict__ qkv,
                                   float* __restrict__ qe,
                                   __bf16* __restrict__ ke,
                                   __bf16* __restrict__ vfT)
{
    long g = (long)blockIdx.x * blockDim.x + threadIdx.x;
    if (g >= (long)BHEADS * NPADK) return;
    int  n  = (int)(g % NPADK);
    long bh = g / NPADK;                  // b*12+h
    __bf16* vcol = vfT + bh * (EDIM * NPADK) + n;   // column n, stride NPADK
    if (n >= NTOK) {                      // zero pad columns (K padding of mid GEMM)
        for (int e = 0; e < EDIM; ++e) vcol[(long)e * NPADK] = (__bf16)0.f;
        return;
    }
    int b = (int)(bh / NHEAD), h = (int)(bh % NHEAD);
    const float* base = qkv + (long)(b * NTOK + n) * (3 * DMODEL) + h * HDIM;
    float*  qrow = qe + (bh * NTOK + n) * EDIM;
    __bf16* krow = ke + (bh * NTOK + n) * EDIM;

    int p = n - 1;
    int py = (p >= 0) ? p / 14 : 0;
    int px = (p >= 0) ? p % 14 : 0;
    float qs = 0.f, ks = 0.f, vs = 0.f;
    for (int j = 0; j < 32; ++j) {
        float cth, sth;
        if (n == 0) { cth = 1.f; sth = 0.f; }
        else {
            float inv = __powf(100.f, -(float)(j & 15) * (1.f / 16.f));
            float ang = ((j < 16) ? (float)py : (float)px) * inv;
            sth = __sinf(ang); cth = __cosf(ang);
        }
        float q0 = base[2 * j], q1 = base[2 * j + 1];
        float r0 = q0 * cth - q1 * sth, r1 = q0 * sth + q1 * cth;
        qs += r0 * r0 + r1 * r1;
        qrow[2 * j] = r0; qrow[2 * j + 1] = r1;
        float k0 = base[DMODEL + 2 * j], k1 = base[DMODEL + 2 * j + 1];
        float t0 = k0 * cth - k1 * sth, t1 = k0 * sth + k1 * cth;
        ks += t0 * t0 + t1 * t1;
        krow[2 * j] = (__bf16)t0; krow[2 * j + 1] = (__bf16)t1;
        float v0 = base[2 * DMODEL + 2 * j], v1 = base[2 * DMODEL + 2 * j + 1];
        vs += v0 * v0 + v1 * v1;
        vcol[(long)(1 + 2 * j) * NPADK] = (__bf16)v0;
        vcol[(long)(2 + 2 * j) * NPADK] = (__bf16)v1;
    }
    qrow[64] = sqrtf(fmaxf(qs + 1.f, 1e-6f));
    krow[64] = (__bf16)(-sqrtf(fmaxf(ks + 1.f, 1e-6f)));
    vcol[0]  = (__bf16)sqrtf(fmaxf(vs + 1.f, 1e-6f));
    for (int i = 65; i < EDIM; ++i) {
        qrow[i] = 0.f; krow[i] = (__bf16)0.f;
        vcol[(long)i * NPADK] = (__bf16)0.f;
    }
}

// softmax over (2C + 2*scores)/scale, in-place, zero K-pad cols
__global__ __launch_bounds__(256)
void softmax_kernel(float* __restrict__ attn, float inv_scale)
{
    __shared__ float sbuf[8];
    long row = blockIdx.x;                // 0..384*197-1
    float* r = attn + row * NPADK;
    int i = threadIdx.x;
    float v = (i < NTOK) ? (2.f + 2.f * r[i]) * inv_scale : -1e30f;
    float m = block_max(v, sbuf);
    float e = (i < NTOK) ? __expf(v - m) : 0.f;
    float s = block_sum(e, sbuf);
    if (i < NPADK) r[i] = (i < NTOK) ? e / s : 0.f;
}

// project mid lorentz vectors, scatter spatial parts per head into aos[T][768]
__global__ void project_mid_kernel(const float* __restrict__ mid, float* __restrict__ aos)
{
    long g = (long)blockIdx.x * blockDim.x + threadIdx.x;
    if (g >= (long)TTOK * NHEAD) return;
    int  h  = (int)(g % NHEAD);
    long bn = g / NHEAD;                  // b*197+n
    int b = (int)(bn / NTOK), n = (int)(bn % NTOK);
    const float* z = mid + ((long)(b * NHEAD + h) * NTOK + n) * EDIM;
    float z0 = z[0]; float ss = 0.f;
    #pragma unroll
    for (int i = 1; i <= 64; ++i) ss += z[i] * z[i];
    float f = rsqrtf(fmaxf(z0 * z0 - ss, 1e-6f));
    float* dst = aos + bn * 768 + h * HDIM;
    #pragma unroll
    for (int i = 0; i < 64; ++i) dst[i] = z[1 + i] * f;
}

__global__ void swiglu_kernel(float* __restrict__ h1, const float* __restrict__ h2, long total)
{
    long g = (long)blockIdx.x * blockDim.x + threadIdx.x;
    if (g >= total) return;
    float a = h1[g];
    h1[g] = (a / (1.f + __expf(-a))) * h2[g];
}

// ---------------------------------------------------------------------------
// Host orchestration
// ---------------------------------------------------------------------------
extern "C" void kernel_launch(void* const* d_in, const int* in_sizes, int n_in,
                              void* d_out, int out_size, void* d_ws, size_t ws_size,
                              hipStream_t stream)
{
    (void)in_sizes; (void)n_in; (void)out_size; (void)ws_size;
    const float* x     = (const float*)d_in[0];
    const float* cls_s = (const float*)d_in[1];
    const float* Wp    = (const float*)d_in[2];
    const float* ln1_g = (const float*)d_in[3];
    const float* ln1_b = (const float*)d_in[4];
    const float* Wq    = (const float*)d_in[5];
    const float* Wk    = (const float*)d_in[6];
    const float* Wv    = (const float*)d_in[7];
    const float* Wo    = (const float*)d_in[8];
    const float* ln2_g = (const float*)d_in[9];
    const float* ln2_b = (const float*)d_in[10];
    const float* W1    = (const float*)d_in[11];
    const float* W2    = (const float*)d_in[12];
    const float* W3    = (const float*)d_in[13];
    const float* wy1   = (const float*)d_in[14];
    const float* wy2   = (const float*)d_in[15];
    const float* lnf_g = (const float*)d_in[16];
    const float* lnf_b = (const float*)d_in[17];
    float* out = (float*)d_out;

    // bump allocator over d_ws
    char* wp_ = (char*)d_ws;
    auto alloc = [&](size_t bytes) -> char* {
        char* r = wp_;
        wp_ += (bytes + 255) & ~(size_t)255;
        return r;
    };
    float*  tok  = (float*)alloc((size_t)TTOK * 768 * 4);
    float*  xn   = (float*)alloc((size_t)TTOK * 768 * 4);
    float*  qkvb = (float*)alloc((size_t)TTOK * 2304 * 4);
    float*  qe   = (float*)alloc((size_t)BHEADS * NTOK * EDIM * 4);
    __bf16* ke   = (__bf16*)alloc((size_t)BHEADS * NTOK * EDIM * 2);
    __bf16* vfT  = (__bf16*)alloc((size_t)BHEADS * EDIM * NPADK * 2);
    float*  attn = (float*)alloc((size_t)BHEADS * NTOK * NPADK * 4);
    float*  mid  = (float*)alloc((size_t)BHEADS * NTOK * EDIM * 4);
    float*  aos  = (float*)alloc((size_t)TTOK * 768 * 4);
    float*  pout = (float*)alloc((size_t)TTOK * 768 * 4);
    float*  h1   = (float*)alloc((size_t)TTOK * DFFN * 4);
    float*  h2   = (float*)alloc((size_t)TTOK * DFFN * 4);
    __bf16* wqkvT = (__bf16*)alloc((size_t)2304 * 768 * 2);   // [2304][768]
    __bf16* woT   = (__bf16*)alloc((size_t)768 * 768 * 2);    // [768][768]
    __bf16* w1T   = (__bf16*)alloc((size_t)DFFN * 768 * 2);   // [2048][768]
    __bf16* w2T   = (__bf16*)alloc((size_t)DFFN * 768 * 2);   // [2048][768]
    __bf16* w3T   = (__bf16*)alloc((size_t)768 * DFFN * 2);   // [768][2048]
    __bf16* wpT   = (__bf16*)alloc((size_t)768 * 1024 * 2);   // [768][1024]
    float*  Ap   = h1;     // alias: patch matrix only used before MLP
    float*  feat = qkvb;   // alias: feat only used before first attention

    auto packT = [&](const float* src, __bf16* dst, int K, int N, int Kp, int Np, int stride) {
        long total = (long)Kp * Np;
        pack_wT_kernel<<<(int)((total + 255) / 256), 256, 0, stream>>>(src, dst, K, N, Kp, Np, stride, total);
    };
    auto gemm = [&](const float* A, const __bf16* Bb, float* C,
                    int Mact, int Nact, int K, int lda, int ldb, int ldc,
                    long sA, long sB, long sC, int batch) {
        dim3 grid((Nact + BN - 1) / BN, (Mact + BM - 1) / BM, batch);
        gemm_wmma_kernel<<<grid, 256, 0, stream>>>(A, Bb, C, Mact, Nact, K, lda, ldb, ldc, sA, sB, sC);
    };

    // ---- patch embedding ----
    build_patches_kernel<<<NBATCH * 196, 256, 0, stream>>>(x, Ap);
    packT(Wp, wpT, 1024, 767, 1024, 768, 1024);
    gemm(Ap, wpT, feat, NBATCH * 196, 768, 1024, 1024, 1024, 768, 0, 0, 0, 1);
    build_tok_kernel<<<TTOK, 256, 0, stream>>>(feat, cls_s, tok);

    const float inv_scale = rsqrtf(768.f);

    for (int l = 0; l < NLAYER; ++l) {
        // pack this layer's weights, transposed (reused staging; stream-ordered)
        packT(Wq + (long)l * 767 * 768, wqkvT + 0 * 768 * 768, 767, 768, 768, 768, 768);
        packT(Wk + (long)l * 767 * 768, wqkvT + 1 * 768 * 768, 767, 768, 768, 768, 768);
        packT(Wv + (long)l * 767 * 768, wqkvT + 2 * 768 * 768, 767, 768, 768, 768, 768);
        packT(Wo + (long)l * 768 * 767, woT, 768, 767, 768, 768, 768);
        packT(W1 + (long)l * 767 * DFFN, w1T, 767, DFFN, 768, DFFN, 768);
        packT(W2 + (long)l * 767 * DFFN, w2T, 767, DFFN, 768, DFFN, 768);
        packT(W3 + (long)l * DFFN * 767, w3T, DFFN, 767, DFFN, 768, DFFN);

        // ---- attention ----
        lorentz_ln_kernel<<<TTOK, 256, 0, stream>>>(tok, xn, ln1_g + (long)l * 767, ln1_b + (long)l * 767, 1);
        gemm(xn, wqkvT, qkvb, TTOK, 2304, 768, 768, 768, 2304, 0, 0, 0, 1);
        build_qkevf_kernel<<<(BHEADS * NPADK + 255) / 256, 256, 0, stream>>>(qkvb, qe, ke, vfT);
        // scores: qe @ ke^T  (batched over b*h)
        gemm(qe, ke, attn, NTOK, NTOK, EDIM, EDIM, EDIM, NPADK,
             (long)NTOK * EDIM, (long)NTOK * EDIM, (long)NTOK * NPADK, BHEADS);
        softmax_kernel<<<BHEADS * NTOK, 256, 0, stream>>>(attn, inv_scale);
        // mid: attn @ vf   (B^T = vfT [96][224])
        gemm(attn, vfT, mid, NTOK, EDIM, NPADK, NPADK, NPADK, EDIM,
             (long)NTOK * NPADK, (long)EDIM * NPADK, (long)NTOK * EDIM, BHEADS);
        project_mid_kernel<<<(int)(((long)TTOK * NHEAD + 255) / 256), 256, 0, stream>>>(mid, aos);
        gemm(aos, woT, pout, TTOK, 768, 768, 768, 768, 768, 0, 0, 0, 1);
        resadd_project_kernel<<<TTOK, 256, 0, stream>>>(tok, pout, wy1, l);

        // ---- MLP ----
        lorentz_ln_kernel<<<TTOK, 256, 0, stream>>>(tok, xn, ln2_g + (long)l * 767, ln2_b + (long)l * 767, 1);
        gemm(xn, w1T, h1, TTOK, DFFN, 768, 768, 768, DFFN, 0, 0, 0, 1);
        gemm(xn, w2T, h2, TTOK, DFFN, 768, 768, 768, DFFN, 0, 0, 0, 1);
        swiglu_kernel<<<(int)(((long)TTOK * DFFN + 255) / 256), 256, 0, stream>>>(h1, h2, (long)TTOK * DFFN);
        gemm(h1, w3T, pout, TTOK, 768, DFFN, DFFN, DFFN, 768, 0, 0, 0, 1);
        resadd_project_kernel<<<TTOK, 256, 0, stream>>>(tok, pout, wy2, l);
    }

    // ---- final lorentz LN -> d_out [B,197,768] ----
    lorentz_ln_kernel<<<TTOK, 256, 0, stream>>>(tok, out, lnf_g, lnf_b, 0);
}
